// MultiHeadSelfAttention_30829275250882
// MI455X (gfx1250) — matrix operand order repaired and checked by reading, not compiled, not run on hardware
//
#include <hip/hip_runtime.h>

typedef __attribute__((ext_vector_type(8)))  float        v8f;
typedef __attribute__((ext_vector_type(16))) __bf16       v16bf;
typedef __attribute__((ext_vector_type(8)))  __bf16       v8bf;
typedef __attribute__((ext_vector_type(4)))  unsigned int u32x4;
typedef __attribute__((ext_vector_type(8)))  int          i32x8;
typedef __attribute__((ext_vector_type(4)))  int          i32x4;

#define B_   2
#define S_   2048
#define D_   1024
#define H_   16
#define DK_  64
#define NT_  (B_ * S_)          // 4096 tokens
#define LDP_ 72                 // padded LDS stride (halfwords)

#if defined(__has_builtin)
#if __has_builtin(__builtin_amdgcn_tensor_load_to_lds)
#define HAVE_TDM 1
#endif
#endif

// ---------- bf16 helpers ----------
__device__ __forceinline__ float bf2f(__bf16 b) {
  unsigned short us = __builtin_bit_cast(unsigned short, b);
  unsigned u = ((unsigned)us) << 16;
  return __builtin_bit_cast(float, u);
}
__device__ __forceinline__ __bf16 f2bf(float f) {
  unsigned u = __builtin_bit_cast(unsigned, f);
  u += 0x7fffu + ((u >> 16) & 1u);            // round-to-nearest-even
  unsigned short us = (unsigned short)(u >> 16);
  return __builtin_bit_cast(__bf16, us);
}

// ---------- WMMA fragment loads (wave32 layouts, CDNA5 ISA 7.12.2) ----------
// A (16x32 bf16): lane holds row = lane%16; elements K = base+0..7 and
// base+16..23 with base = (lane<16 ? 0 : 8). Caller passes p = row + base.
__device__ __forceinline__ v16bf ld_afrag(const __bf16* p) {
  v8bf lo = *(const v8bf*)(p);
  v8bf hi = *(const v8bf*)(p + 16);
  return __builtin_shufflevector(lo, hi, 0,1,2,3,4,5,6,7,8,9,10,11,12,13,14,15);
}
// B (32x16 bf16): lane holds col = lane%16; elements K = base+0..7 and
// base+16..23 with base = (lane<16 ? 0 : 8). Caller passes p = col + base.
__device__ __forceinline__ v16bf ld_bfrag(const __bf16* p) {
  v8bf lo = *(const v8bf*)(p);
  v8bf hi = *(const v8bf*)(p + 16);
  return __builtin_shufflevector(lo, hi, 0,1,2,3,4,5,6,7,8,9,10,11,12,13,14,15);
}
__device__ __forceinline__ v8f wmma_bf16(v16bf a, v16bf b, v8f c) {
  return __builtin_amdgcn_wmma_f32_16x16x32_bf16(false, a, false, b,
                                                 (short)0, c, false, false);
}

#ifdef HAVE_TDM
// ---------- Tensor Data Mover: 2D bf16 tile -> LDS (ISA ch.8 D# layout) ----
// tile_w elems per row (must be 64 here: pad fields assume 32-DWORD rows with
// 4-DWORD padding -> LDS row stride 72 halfwords), tile_h rows,
// row_stride/elems between tensor rows in memory.
__device__ __forceinline__ void tdm_load_2d(unsigned lds_byte_off,
                                            const void* gptr,
                                            unsigned tile_w, unsigned tile_h,
                                            unsigned row_stride,
                                            unsigned tensor_rows) {
  unsigned long long ga = (unsigned long long)gptr;
  u32x4 g0 = {
      1u,                                            // count=1, user mode
      lds_byte_off,                                  // lds_addr[63:32]
      (unsigned)(ga & 0xffffffffu),                  // global_addr[95:64]
      (unsigned)((ga >> 32) & 0x1ffffffu) | (2u << 30)  // addr hi | type=2
  };
  i32x8 g1 = {
      (int)((1u << 16) | (1u << 20) | (4u << 22) | (3u << 25)),
      //     data_size=2B   pad_en     32-DWORD int.  4-DWORD pad
      (int)((row_stride & 0xffffu) << 16),           // tensor_dim0[15:0]
      (int)(((row_stride >> 16) & 0xffffu) |
            ((tensor_rows & 0xffffu) << 16)),        // dim0 hi | dim1 lo
      (int)(((tensor_rows >> 16) & 0xffffu) |
            ((tile_w & 0xffffu) << 16)),             // dim1 hi | tile_dim0
      (int)(tile_h & 0xffffu),                       // tile_dim1 | tile_dim2=0
      (int)row_stride,                               // tensor_dim0_stride lo32
      0,                                             // stride hi | dim1_stride lo
      0
  };
  i32x4 z4 = {0, 0, 0, 0};                           // 2D: groups 2/3 unused
  i32x8 z8 = {0, 0, 0, 0, 0, 0, 0, 0};
  __builtin_amdgcn_tensor_load_to_lds(g0, g1, z4, z4, z8, 0);
}
__device__ __forceinline__ unsigned lds_offset_of(const void* p) {
  return (unsigned)(unsigned long long)p;            // LDS aperture: low 32 bits
}
#endif

// ---------- fp32 -> bf16 conversion ----------
__global__ __launch_bounds__(256) void cvt_kernel(const float* __restrict__ src,
                                                  __bf16* __restrict__ dst, int n) {
  int i = blockIdx.x * 256 + threadIdx.x;
  if (i < n) dst[i] = f2bf(src[i]);
}

// ---------- RoPE (in place, [B,H,S,Dk] bf16), optional scale fold ----------
__global__ __launch_bounds__(256) void rope_kernel(__bf16* __restrict__ t,
                                                   const int* __restrict__ pos,
                                                   float scale) {
  int idx = blockIdx.x * 256 + threadIdx.x;            // pair index
  const int NPAIR = B_ * H_ * S_ * (DK_ / 2);
  if (idx >= NPAIR) return;
  int dp = idx % (DK_ / 2);
  int s  = (idx / (DK_ / 2)) % S_;
  int bh = idx / ((DK_ / 2) * S_);
  int b  = bh / H_;
  float p    = (float)pos[b * S_ + s];
  float freq = __expf(-((float)(2 * dp) / (float)DK_) * 9.210340371976184f);
  float ang  = p * freq;
  float sn, cs;
  __sincosf(ang, &sn, &cs);
  __bf16* e = t + (size_t)idx * 2;
  float xe = bf2f(e[0]), xo = bf2f(e[1]);
  e[0] = f2bf((xe * cs - xo * sn) * scale);
  e[1] = f2bf((xe * sn + xo * cs) * scale);
}

// ---------- GEMM: Y[M,N] = A[M,K] @ W[N,K]^T, bf16 in, f32 accum ----------
// mode 0: write bf16 head-major [B,H,S,Dk];  mode 1: write fp32 row-major.
// TDM path: double-buffered LDS tiles; wave 0 issues next stage's DMA pair,
// waits TENSORcnt<=2 (previous pair landed, in-order completion), barrier,
// all waves compute while the TDM engine fills the other buffer.
__global__ __launch_bounds__(256) void gemm_kernel(const __bf16* __restrict__ A,
                                                   const __bf16* __restrict__ W,
                                                   __bf16* __restrict__ outh,
                                                   float* __restrict__ outf,
                                                   int mode, int mrows) {
#ifdef HAVE_TDM
  __shared__ __align__(16) __bf16 sA[2][128 * LDP_];
  __shared__ __align__(16) __bf16 sB[2][128 * LDP_];
#else
  __shared__ __align__(16) __bf16 sA[1][128 * LDP_];
  __shared__ __align__(16) __bf16 sB[1][128 * LDP_];
#endif
  const int K = D_;
  const unsigned tid  = threadIdx.x;
  const unsigned lane = tid & 31u, w = tid >> 5;
  const unsigned lm   = lane & 15u, half = lane >> 4;
  const int m0 = blockIdx.x * 128;
  const int n0 = blockIdx.y * 128;
  const int wm = (int)(w & 1u);        // 2 wave-rows  of 64
  const int wn = (int)(w >> 1);        // 4 wave-cols  of 32

  v8f acc[4][2];
#pragma unroll
  for (int tm = 0; tm < 4; ++tm)
#pragma unroll
    for (int tn = 0; tn < 2; ++tn)
      acc[tm][tn] = v8f{0.f, 0.f, 0.f, 0.f, 0.f, 0.f, 0.f, 0.f};

#ifdef HAVE_TDM
  if (w == 0) {                                   // prologue: stage 0
    tdm_load_2d(lds_offset_of(sA[0]), A + (size_t)m0 * K, 64u, 128u,
                (unsigned)K, (unsigned)mrows);
    tdm_load_2d(lds_offset_of(sB[0]), W + (size_t)n0 * K, 64u, 128u,
                (unsigned)K, (unsigned)D_);
  }
#endif

  for (int kb = 0; kb < K; kb += 64) {
#ifdef HAVE_TDM
    const int ib = (kb >> 6) & 1;
    if (w == 0) {
      if (kb + 64 < K) {                          // issue stage kb+64 into other buf
        tdm_load_2d(lds_offset_of(sA[ib ^ 1]), A + (size_t)m0 * K + kb + 64,
                    64u, 128u, (unsigned)K, (unsigned)mrows);
        tdm_load_2d(lds_offset_of(sB[ib ^ 1]), W + (size_t)n0 * K + kb + 64,
                    64u, 128u, (unsigned)K, (unsigned)D_);
        __builtin_amdgcn_s_wait_tensorcnt(2);     // stage kb's pair has landed
      } else {
        __builtin_amdgcn_s_wait_tensorcnt(0);
      }
    }
    __syncthreads();                              // publish buf[ib] to all waves
#else
    const int ib = 0;
    (void)mrows;
#pragma unroll
    for (int i = 0; i < 4; ++i) {                 // 1024 16B chunks per tile
      int idx = (int)tid + i * 256;
      int r = idx >> 3, cc = idx & 7;
      *(v8bf*)(sA[0] + r * LDP_ + cc * 8) =
          *(const v8bf*)(A + (size_t)(m0 + r) * K + kb + cc * 8);
      *(v8bf*)(sB[0] + r * LDP_ + cc * 8) =
          *(const v8bf*)(W + (size_t)(n0 + r) * K + kb + cc * 8);
    }
    __syncthreads();
#endif
#pragma unroll
    for (int kk = 0; kk < 64; kk += 32) {
      v16bf af[4], bf[2];
#pragma unroll
      for (int tm = 0; tm < 4; ++tm)
        af[tm] = ld_afrag(sA[ib] + (wm * 64 + tm * 16 + (int)lm) * LDP_ + kk + (int)half * 8);
#pragma unroll
      for (int tn = 0; tn < 2; ++tn)
        bf[tn] = ld_bfrag(sB[ib] + (wn * 32 + tn * 16 + (int)lm) * LDP_ + kk + (int)half * 8);
#pragma unroll
      for (int tm = 0; tm < 4; ++tm)
#pragma unroll
        for (int tn = 0; tn < 2; ++tn)
          acc[tm][tn] = wmma_bf16(af[tm], bf[tn], acc[tm][tn]);
    }
    __syncthreads();                              // reads of buf[ib] complete
  }

#pragma unroll
  for (int tm = 0; tm < 4; ++tm)
#pragma unroll
    for (int tn = 0; tn < 2; ++tn)
#pragma unroll
      for (int v = 0; v < 8; ++v) {
        int gm = m0 + wm * 64 + tm * 16 + v + (int)half * 8;
        int gn = n0 + wn * 32 + tn * 16 + (int)lm;
        float val = acc[tm][tn][v];
        if (mode == 0) {
          int b = gm >> 11, s = gm & 2047, h = gn >> 6, d = gn & 63;
          outh[(((size_t)(b * H_ + h)) * S_ + s) * DK_ + d] = f2bf(val);
        } else {
          outf[(size_t)gm * D_ + gn] = val;
        }
      }
}

// ---------- Flash attention: per (b,h, 128-query tile) ----------
__global__ __launch_bounds__(256) void attn_kernel(const __bf16* __restrict__ Qb,
                                                   const __bf16* __restrict__ Kb,
                                                   const __bf16* __restrict__ Vb,
                                                   __bf16* __restrict__ Ob) {
  __shared__ __align__(16) __bf16 sK [64 * LDP_];       // [key][d]
  __shared__ __align__(16) __bf16 sVt[64 * LDP_];       // [d][key] (transposed)
  __shared__ __align__(16) __bf16 sP [8 * 16 * LDP_];   // per-wave P tiles

  const int qt = blockIdx.x;
  const int bh = blockIdx.y;
  const int b  = bh >> 4, h = bh & 15;
  const unsigned tid  = threadIdx.x;
  const unsigned lane = tid & 31u, w = tid >> 5;
  const unsigned lm   = lane & 15u, half = lane >> 4;
  const int q0 = qt * 128 + (int)w * 16;                 // wave's first query

  const __bf16* Qp = Qb + (size_t)bh * S_ * DK_;
  const __bf16* Kp = Kb + (size_t)bh * S_ * DK_;
  const __bf16* Vp = Vb + (size_t)bh * S_ * DK_;

  // Q fragments held in registers (2 k-steps over Dk=64)
  const __bf16* qrow = Qp + (size_t)(q0 + (int)lm) * DK_;
  v16bf qa[2];
  qa[0] = ld_afrag(qrow + 0  + (int)half * 8);
  qa[1] = ld_afrag(qrow + 32 + (int)half * 8);

  float mrow[8], lrow[8];
  v8f oacc[4];
#pragma unroll
  for (int v = 0; v < 8; ++v) { mrow[v] = -1e30f; lrow[v] = 0.f; }
#pragma unroll
  for (int dt = 0; dt < 4; ++dt) oacc[dt] = v8f{0.f,0.f,0.f,0.f,0.f,0.f,0.f,0.f};

  const int nkt = qt * 2 + 2;                            // causal key chunks of 64
  for (int kt = 0; kt < nkt; ++kt) {
    const int k0g = kt * 64;
    // stage K chunk (row-major) and V chunk (transposed)
#pragma unroll
    for (int i = 0; i < 2; ++i) {
      int idx = (int)tid + i * 256;                      // 512 chunks of 8
      int r = idx >> 3, cc = idx & 7;
      *(v8bf*)(sK + r * LDP_ + cc * 8) =
          *(const v8bf*)(Kp + (size_t)(k0g + r) * DK_ + cc * 8);
      v8bf vv = *(const v8bf*)(Vp + (size_t)(k0g + r) * DK_ + cc * 8);
#pragma unroll
      for (int j = 0; j < 8; ++j) sVt[(cc * 8 + j) * LDP_ + r] = vv[j];
    }
    __syncthreads();

    if (k0g <= q0) {                                     // wave-uniform causal skip
      // S = Q @ K^T  (4 key sub-tiles x 2 k-steps)
      v8f sc[4];
#pragma unroll
      for (int nt = 0; nt < 4; ++nt) {
        sc[nt] = v8f{0.f,0.f,0.f,0.f,0.f,0.f,0.f,0.f};
#pragma unroll
        for (int ks = 0; ks < 2; ++ks) {
          v16bf kb = ld_bfrag(sK + (nt * 16 + (int)lm) * LDP_ + ks * 32 + (int)half * 8);
          sc[nt] = wmma_bf16(qa[ks], kb, sc[nt]);
        }
      }
      // causal mask + running max
      float nmax[8];
#pragma unroll
      for (int v = 0; v < 8; ++v) nmax[v] = mrow[v];
#pragma unroll
      for (int nt = 0; nt < 4; ++nt)
#pragma unroll
        for (int v = 0; v < 8; ++v) {
          int jg = k0g + nt * 16 + (int)lm;
          int qg = q0 + v + (int)half * 8;
          float sv = (jg <= qg) ? sc[nt][v] : -1e30f;
          sc[nt][v] = sv;
          nmax[v] = fmaxf(nmax[v], sv);
        }
#pragma unroll
      for (int dxor = 1; dxor < 16; dxor <<= 1)
#pragma unroll
        for (int v = 0; v < 8; ++v)
          nmax[v] = fmaxf(nmax[v], __shfl_xor(nmax[v], dxor, 32));

      float corr[8], rs[8];
#pragma unroll
      for (int v = 0; v < 8; ++v) { corr[v] = __expf(mrow[v] - nmax[v]); rs[v] = 0.f; }

      __bf16* spw = sP + (int)w * 16 * LDP_;
#pragma unroll
      for (int nt = 0; nt < 4; ++nt)
#pragma unroll
        for (int v = 0; v < 8; ++v) {
          float p = __expf(sc[nt][v] - nmax[v]);
          rs[v] += p;
          spw[(v + (int)half * 8) * LDP_ + nt * 16 + (int)lm] = f2bf(p);
        }
#pragma unroll
      for (int dxor = 1; dxor < 16; dxor <<= 1)
#pragma unroll
        for (int v = 0; v < 8; ++v)
          rs[v] += __shfl_xor(rs[v], dxor, 32);
#pragma unroll
      for (int v = 0; v < 8; ++v) {
        lrow[v] = lrow[v] * corr[v] + rs[v];
        mrow[v] = nmax[v];
      }
#pragma unroll
      for (int dt = 0; dt < 4; ++dt)
#pragma unroll
        for (int v = 0; v < 8; ++v) oacc[dt][v] *= corr[v];

      // O += P @ V   (per-wave LDS round-trip; DS ops are in-order per wave)
#pragma unroll
      for (int ks = 0; ks < 2; ++ks) {
        v16bf pa = ld_afrag(spw + (int)lm * LDP_ + ks * 32 + (int)half * 8);
#pragma unroll
        for (int dt = 0; dt < 4; ++dt) {
          v16bf vb = ld_bfrag(sVt + (dt * 16 + (int)lm) * LDP_ + ks * 32 + (int)half * 8);
          oacc[dt] = wmma_bf16(pa, vb, oacc[dt]);
        }
      }
    }
    __syncthreads();
  }

  float inv[8];
#pragma unroll
  for (int v = 0; v < 8; ++v) inv[v] = 1.0f / fmaxf(lrow[v], 1e-30f);
#pragma unroll
  for (int dt = 0; dt < 4; ++dt)
#pragma unroll
    for (int v = 0; v < 8; ++v) {
      int qg = q0 + v + (int)half * 8;
      size_t off = ((size_t)b * S_ + qg) * D_ + h * DK_ + dt * 16 + (int)lm;
      Ob[off] = f2bf(oacc[dt][v] * inv[v]);
    }
}

// ---------- host launch ----------
extern "C" void kernel_launch(void* const* d_in, const int* in_sizes, int n_in,
                              void* d_out, int out_size, void* d_ws, size_t ws_size,
                              hipStream_t stream) {
  (void)in_sizes; (void)n_in; (void)out_size; (void)ws_size;
  const float* x  = (const float*)d_in[0];
  const float* Wq = (const float*)d_in[1];
  const float* Wk = (const float*)d_in[2];
  const float* Wv = (const float*)d_in[3];
  const float* Wo = (const float*)d_in[4];
  const int* pos  = (const int*)d_in[5];

  char* ws = (char*)d_ws;
  const size_t MB = 1u << 20;
  __bf16* xb  = (__bf16*)(ws + 0 * MB);    // [4096,1024]     8 MB
  __bf16* Wqb = (__bf16*)(ws + 8 * MB);    // [1024,1024]     2 MB
  __bf16* Wkb = (__bf16*)(ws + 10 * MB);
  __bf16* Wvb = (__bf16*)(ws + 12 * MB);
  __bf16* Wob = (__bf16*)(ws + 14 * MB);
  __bf16* Qb  = (__bf16*)(ws + 16 * MB);   // [B,H,S,Dk]      8 MB
  __bf16* Kb  = (__bf16*)(ws + 24 * MB);
  __bf16* Vb  = (__bf16*)(ws + 32 * MB);
  __bf16* Ob  = (__bf16*)(ws + 40 * MB);   // [B,S,D]         8 MB

  const int NX = NT_ * D_;                 // 4,194,304
  const int NW = D_ * D_;                  // 1,048,576
  cvt_kernel<<<(NX + 255) / 256, 256, 0, stream>>>(x,  xb,  NX);
  cvt_kernel<<<(NW + 255) / 256, 256, 0, stream>>>(Wq, Wqb, NW);
  cvt_kernel<<<(NW + 255) / 256, 256, 0, stream>>>(Wk, Wkb, NW);
  cvt_kernel<<<(NW + 255) / 256, 256, 0, stream>>>(Wv, Wvb, NW);
  cvt_kernel<<<(NW + 255) / 256, 256, 0, stream>>>(Wo, Wob, NW);

  dim3 gg(NT_ / 128, D_ / 128);            // (32, 8)
  gemm_kernel<<<gg, 256, 0, stream>>>(xb, Wqb, Qb, nullptr, 0, NT_);
  gemm_kernel<<<gg, 256, 0, stream>>>(xb, Wkb, Kb, nullptr, 0, NT_);
  gemm_kernel<<<gg, 256, 0, stream>>>(xb, Wvb, Vb, nullptr, 0, NT_);

  const int NPAIR = B_ * H_ * S_ * (DK_ / 2);   // 2,097,152
  rope_kernel<<<(NPAIR + 255) / 256, 256, 0, stream>>>(Qb, pos, 0.125f);  // 1/sqrt(Dk) folded
  rope_kernel<<<(NPAIR + 255) / 256, 256, 0, stream>>>(Kb, pos, 1.0f);

  attn_kernel<<<dim3(S_ / 128, B_ * H_), 256, 0, stream>>>(Qb, Kb, Vb, Ob);

  gemm_kernel<<<gg, 256, 0, stream>>>(Ob, Wob, nullptr, (float*)d_out, 1, NT_);
}
